// WindowAttentionV2_62680752718178
// MI455X (gfx1250) — compile-verified, matrix operand order checked by reading, not run on hardware
//
#include <hip/hip_runtime.h>
#include <hip/hip_bf16.h>

typedef __attribute__((ext_vector_type(16))) __bf16 bf16x16;
typedef __attribute__((ext_vector_type(8)))  float  floatx8;
typedef unsigned short ushort_t;
typedef unsigned int   uint_t;

#define WIN_N 64
#define CHN   512
#define NH    16
#define HD    32
#define SLP   68                           // padded f32 stride for sL
#define LOG_SCALE_MAX 4.605170185988091f   // log(1/0.01)

// Native bf16 conversions (v_cvt_pk_bf16_f32 on gfx1250).
__device__ __forceinline__ ushort_t f2bf(float f) {
  union { __bf16 h; ushort_t u; } x; x.h = (__bf16)f; return x.u;
}
__device__ __forceinline__ float bf2f(ushort_t u) {
  union { ushort_t u; __bf16 h; } x; x.u = u; return (float)x.h;
}

union Frag  { bf16x16 v; uint4 q[2]; };
union Acc   { floatx8 v; float f[8]; };
union Pack8 { uint4 q; ushort_t us[8]; };
union Pack4 { uint2 u; ushort_t us[4]; };

// CDNA5 async global->LDS copy (ASYNCcnt-tracked, bypasses VGPRs).
// lds_off is the byte offset of the destination within this workgroup's LDS.
__device__ __forceinline__ void async_ld_b128(uint_t lds_off, const void* g) {
  asm volatile("global_load_async_to_lds_b128 %0, %1, off"
               :: "v"(lds_off), "v"((unsigned long long)(uintptr_t)g)
               : "memory");
}
__device__ __forceinline__ void wait_async0() {
  asm volatile("s_wait_asynccnt 0x0" ::: "memory");
}

// Row-major fragment load (A, and B stored as [N][K]).
// Per lane: row = lo; two contiguous 16B K-chunks at 8*half and 16+8*half.
__device__ __forceinline__ Frag ld_frag(const ushort_t* base, int row, int kbase,
                                        int stride, int half) {
  Frag f;
  const ushort_t* p = base + (size_t)row * stride + kbase + half * 8;
  f.q[0] = *(const uint4*)(p);
  f.q[1] = *(const uint4*)(p + 16);
  return f;
}

// ---------------- Preprocessing kernels ----------------

__global__ void prep_weights_kernel(const float* __restrict__ wqkv,
                                    const float* __restrict__ projw,
                                    ushort_t* __restrict__ wqkvT,
                                    ushort_t* __restrict__ projwT) {
  int idx = blockIdx.x * blockDim.x + threadIdx.x;
  int stride = gridDim.x * blockDim.x;
  for (int i = idx; i < 1536 * 512; i += stride) {
    int c = i >> 9, k = i & 511;                       // wqkvT[c][k] = wqkv[k][c]
    wqkvT[i] = f2bf(wqkv[(size_t)k * 1536 + c]);
  }
  for (int i = idx; i < 512 * 512; i += stride) {
    int n = i >> 9, k = i & 511;                       // projwT[n][k] = projw[k][n]
    projwT[i] = f2bf(projw[(size_t)k * 512 + n]);
  }
}

// CPB MLP: table[p][h] = (relu(rel[p]·W1 + b1)) · W2 , p in [0,225)
__global__ __launch_bounds__(256) void cpb_table_kernel(
    const float* __restrict__ rel, const float* __restrict__ w1,
    const float* __restrict__ b1, const float* __restrict__ w2,
    float* __restrict__ table) {
  __shared__ float red[256 * 16];
  int p = blockIdx.x;
  float t0 = rel[p * 2 + 0];
  float t1 = rel[p * 2 + 1];
  int t = threadIdx.x;
  float acc[16];
#pragma unroll
  for (int h = 0; h < 16; ++h) acc[h] = 0.f;
  for (int j = t; j < 512; j += 256) {
    float hdn = fmaxf(0.f, t0 * w1[j] + t1 * w1[512 + j] + b1[j]);
#pragma unroll
    for (int h = 0; h < 16; ++h) acc[h] += hdn * w2[j * 16 + h];
  }
#pragma unroll
  for (int h = 0; h < 16; ++h) red[t * 16 + h] = acc[h];
  __syncthreads();
  if (t < 16) {
    float s = 0.f;
    for (int j = 0; j < 256; ++j) s += red[j * 16 + t];
    table[p * 16 + t] = s;
  }
}

// biasTab[h][i][j] = 16*sigmoid(table[relidx(i,j)][h])
__global__ void bias_expand_kernel(const float* __restrict__ table,
                                   float* __restrict__ biasTab) {
  int idx = blockIdx.x * blockDim.x + threadIdx.x;
  if (idx >= 16 * 64 * 64) return;
  int h = idx >> 12;
  int i = (idx >> 6) & 63;
  int j = idx & 63;
  int yi = i >> 3, xi = i & 7, yj = j >> 3, xj = j & 7;
  int p = (yi - yj + 7) * 15 + (xi - xj + 7);
  float v = table[p * 16 + h];
  biasTab[idx] = 16.f / (1.f + __expf(-v));
}

// ---------------- Fused window-attention kernel ----------------
// One workgroup (256 threads = 8 wave32) per window.
__global__ __launch_bounds__(256) void win_attn_main(
    const float* __restrict__ x, const float* __restrict__ mask,
    const ushort_t* __restrict__ wqkvT, const float* __restrict__ q_bias,
    const float* __restrict__ v_bias, const float* __restrict__ logit_scale,
    const ushort_t* __restrict__ projwT, const float* __restrict__ proj_b,
    const float* __restrict__ biasTab, float* __restrict__ out) {
  extern __shared__ char smem[];
  ushort_t* xs   = (ushort_t*)smem;               // [64][512] bf16 input
  ushort_t* outs = xs + WIN_N * CHN;              // [64][512] bf16 attn output
  ushort_t* qh   = outs + WIN_N * CHN;            // [64][32]  per-head q (norm*scale)
  ushort_t* kh   = qh + WIN_N * HD;               // [64][32]  per-head k (norm)
  ushort_t* vhT  = kh + WIN_N * HD;               // [32][64]  per-head v, transposed
  float*    sL   = (float*)(vhT + HD * WIN_N);    // [64][68]  logits (padded)
  ushort_t* pL   = (ushort_t*)(sL + WIN_N * SLP); // [64][64]  probs bf16
  float*    sLb  = (float*)(pL + WIN_N * WIN_N);  // [64][64]  CPB bias (async)
  float*    sLm  = sLb + WIN_N * WIN_N;           // [64][64]  mask (async, once)

  const int tid  = threadIdx.x;
  const int wid  = tid >> 5;
  const int lane = tid & 31;
  const int lo   = lane & 15;
  const int half = lane >> 4;
  const int bwin = blockIdx.x;

  const uint_t sLb_off = (uint_t)((char*)sLb - smem);
  const uint_t sLm_off = (uint_t)((char*)sLm - smem);

  // Async-stage the window mask into LDS once (overlaps Phase 1).
  const float* maskW = mask + (size_t)(bwin & 255) * WIN_N * WIN_N;
  for (int j = tid; j < (WIN_N * WIN_N) / 4; j += 256)
    async_ld_b128(sLm_off + j * 16, maskW + j * 4);

  // Phase 1: stage x window as bf16 in LDS (b128 loads, packed b64 stores)
  const float* xw = x + (size_t)bwin * WIN_N * CHN;
  for (int i = tid * 4; i < WIN_N * CHN; i += 256 * 4) {
    float4 f = *(const float4*)(xw + i);
    Pack4 pk;
    pk.us[0] = f2bf(f.x); pk.us[1] = f2bf(f.y);
    pk.us[2] = f2bf(f.z); pk.us[3] = f2bf(f.w);
    *(uint2*)(xs + i) = pk.u;
  }
  __syncthreads();

  for (int h = 0; h < NH; ++h) {
    const float scale_h = __expf(fminf(logit_scale[h], LOG_SCALE_MAX));
    const float* biasH = biasTab + (size_t)h * WIN_N * WIN_N;

    // Kick off async copy of this head's CPB bias tile; it lands in LDS
    // while the QKV GEMM below is running.
    for (int j = tid; j < (WIN_N * WIN_N) / 4; j += 256)
      async_ld_b128(sLb_off + j * 16, biasH + j * 4);

    // Phase A: per-head QKV GEMM -> q/k/v [64,32]; 24 tiles over 8 waves
#pragma unroll
    for (int r = 0; r < 3; ++r) {
      int t = wid + 8 * r;
      int mt = t & 3;
      int cq = t >> 2;
      int which = cq >> 1;      // 0=q 1=k 2=v
      int nt = cq & 1;
      int cbase = which * CHN + h * HD + nt * 16;
      float bc = 0.f;
      if (which == 0) bc = q_bias[h * HD + nt * 16 + lo];
      else if (which == 2) bc = v_bias[h * HD + nt * 16 + lo];
      Acc acc;
#pragma unroll
      for (int e = 0; e < 8; ++e) acc.f[e] = bc;
#pragma unroll
      for (int kk = 0; kk < CHN; kk += 32) {
        Frag a = ld_frag(xs, mt * 16 + lo, kk, CHN, half);
        Frag b = ld_frag(wqkvT, cbase + lo, kk, CHN, half);
        acc.v = __builtin_amdgcn_wmma_f32_16x16x32_bf16(
            false, a.v, false, b.v, (short)0, acc.v, false, false);
      }
      if (which == 0) {
#pragma unroll
        for (int e = 0; e < 8; ++e)
          qh[(mt * 16 + e + 8 * half) * HD + nt * 16 + lo] = f2bf(acc.f[e]);
      } else if (which == 1) {
#pragma unroll
        for (int e = 0; e < 8; ++e)
          kh[(mt * 16 + e + 8 * half) * HD + nt * 16 + lo] = f2bf(acc.f[e]);
      } else {  // v transposed: per lane the 8 tokens are contiguous -> b128
        Pack8 pk;
#pragma unroll
        for (int e = 0; e < 8; ++e) pk.us[e] = f2bf(acc.f[e]);
        *(uint4*)(vhT + (nt * 16 + lo) * WIN_N + mt * 16 + 8 * half) = pk.q;
      }
    }
    __syncthreads();

    // Phase B: normalize q/k rows (cosine attention), fold scale into q
    if (tid < 128) {
      int r = tid & 63;
      ushort_t* rp = (tid < 64 ? qh : kh) + r * HD;
      float ss = 0.f;
#pragma unroll
      for (int j = 0; j < HD; ++j) { float f = bf2f(rp[j]); ss += f * f; }
      float inv = rsqrtf(ss + 1e-12f);
      if (tid < 64) inv *= scale_h;
#pragma unroll
      for (int j = 0; j < HD; ++j) rp[j] = f2bf(bf2f(rp[j]) * inv);
    }
    wait_async0();          // bias (and first-iteration mask) landed in LDS
    __syncthreads();

    // Phase C: S = qn @ kn^T + bias + mask, 16 tiles, single K=32 step
#pragma unroll
    for (int r = 0; r < 2; ++r) {
      int t = wid + 8 * r;
      int mt = t & 3, nt = t >> 2;
      Acc acc;
#pragma unroll
      for (int e = 0; e < 8; ++e) {
        int m = mt * 16 + e + 8 * half;
        int n = nt * 16 + lo;
        acc.f[e] = sLb[m * 64 + n] + sLm[m * 64 + n];
      }
      Frag a = ld_frag(qh, mt * 16 + lo, 0, HD, half);
      Frag b = ld_frag(kh, nt * 16 + lo, 0, HD, half);
      acc.v = __builtin_amdgcn_wmma_f32_16x16x32_bf16(
          false, a.v, false, b.v, (short)0, acc.v, false, false);
#pragma unroll
      for (int e = 0; e < 8; ++e)
        sL[(mt * 16 + e + 8 * half) * SLP + nt * 16 + lo] = acc.f[e];
    }
    __syncthreads();

    // Phase D: row softmax; packed bf16 prob stores (b128)
    if (tid < 64) {
      float* rp = sL + tid * SLP;
      float mx = -3.0e38f;
#pragma unroll
      for (int j = 0; j < 64; ++j) mx = fmaxf(mx, rp[j]);
      float s = 0.f;
#pragma unroll
      for (int j = 0; j < 64; ++j) { float e_ = __expf(rp[j] - mx); rp[j] = e_; s += e_; }
      float inv = 1.f / s;
      ushort_t* pp = pL + tid * 64;
#pragma unroll
      for (int jj = 0; jj < 64; jj += 8) {
        Pack8 pk;
#pragma unroll
        for (int u = 0; u < 8; ++u) pk.us[u] = f2bf(rp[jj + u] * inv);
        *(uint4*)(pp + jj) = pk.q;
      }
    }
    __syncthreads();

    // Phase E: O_h = P @ V_h, 8 tiles (one per wave), K loop of 2
    {
      int mt = wid & 3, nt = wid >> 2;
      Acc acc;
#pragma unroll
      for (int e = 0; e < 8; ++e) acc.f[e] = 0.f;
#pragma unroll
      for (int kk = 0; kk < 64; kk += 32) {
        Frag a = ld_frag(pL, mt * 16 + lo, kk, 64, half);
        Frag b = ld_frag(vhT, nt * 16 + lo, kk, 64, half);
        acc.v = __builtin_amdgcn_wmma_f32_16x16x32_bf16(
            false, a.v, false, b.v, (short)0, acc.v, false, false);
      }
#pragma unroll
      for (int e = 0; e < 8; ++e)
        outs[(mt * 16 + e + 8 * half) * CHN + h * HD + nt * 16 + lo] = f2bf(acc.f[e]);
    }
    __syncthreads();
  }

  // Phase F: projection GEMM [64,512] x [512,512] + proj_b -> global fp32
  float* op = out + (size_t)bwin * WIN_N * CHN;
#pragma unroll 2
  for (int r = 0; r < 16; ++r) {
    int t = wid + 8 * r;
    int mt = t & 3, nt = t >> 2;
    float bc = proj_b[nt * 16 + lo];
    Acc acc;
#pragma unroll
    for (int e = 0; e < 8; ++e) acc.f[e] = bc;
#pragma unroll
    for (int kk = 0; kk < CHN; kk += 32) {
      Frag a = ld_frag(outs, mt * 16 + lo, kk, CHN, half);
      Frag b = ld_frag(projwT, nt * 16 + lo, kk, CHN, half);
      acc.v = __builtin_amdgcn_wmma_f32_16x16x32_bf16(
          false, a.v, false, b.v, (short)0, acc.v, false, false);
    }
#pragma unroll
    for (int e = 0; e < 8; ++e)
      op[(mt * 16 + e + 8 * half) * CHN + nt * 16 + lo] = acc.f[e];
  }
}

extern "C" void kernel_launch(void* const* d_in, const int* in_sizes, int n_in,
                              void* d_out, int out_size, void* d_ws, size_t ws_size,
                              hipStream_t stream) {
  (void)in_sizes; (void)n_in; (void)out_size; (void)ws_size;
  const float* x      = (const float*)d_in[0];
  const float* mask   = (const float*)d_in[1];
  const float* rel    = (const float*)d_in[2];
  const float* wqkv   = (const float*)d_in[3];
  const float* q_bias = (const float*)d_in[4];
  const float* v_bias = (const float*)d_in[5];
  const float* lscale = (const float*)d_in[6];
  const float* cpb_w1 = (const float*)d_in[7];
  const float* cpb_b1 = (const float*)d_in[8];
  const float* cpb_w2 = (const float*)d_in[9];
  const float* projw  = (const float*)d_in[10];
  const float* proj_b = (const float*)d_in[11];
  float* out = (float*)d_out;

  // Workspace layout (~2.4 MB): bf16 transposed weights + CPB tables
  ushort_t* wqkvT  = (ushort_t*)d_ws;
  ushort_t* projwT = wqkvT + 1536 * 512;
  float* table   = (float*)((char*)d_ws + (size_t)(1536 * 512 + 512 * 512) * 2);
  float* biasTab = table + 225 * 16;

  prep_weights_kernel<<<2048, 256, 0, stream>>>(wqkv, projw, wqkvT, projwT);
  cpb_table_kernel<<<225, 256, 0, stream>>>(rel, cpb_w1, cpb_b1, cpb_w2, table);
  bias_expand_kernel<<<256, 256, 0, stream>>>(table, biasTab);

  // Dynamic LDS: xs+outs(128K) qh/kh/vhT(12K) sL(17K) pL(8K) sLb/sLm(32K)
  const int smem_bytes = (WIN_N * CHN) * 2 * 2 + (WIN_N * HD) * 2 * 3 +
                         (WIN_N * SLP) * 4 + (WIN_N * WIN_N) * 2 +
                         (WIN_N * WIN_N) * 4 * 2;   // 201728 B
  hipFuncSetAttribute((const void*)win_attn_main,
                      hipFuncAttributeMaxDynamicSharedMemorySize, smem_bytes);
  win_attn_main<<<2048, 256, smem_bytes, stream>>>(
      x, mask, wqkvT, q_bias, v_bias, lscale, projwT, proj_b, biasTab, out);
}
